// CRFDecode_68178310857019
// MI455X (gfx1250) — matrix operand order repaired and checked by reading, not compiled, not run on hardware
//
#include <hip/hip_runtime.h>
#include <hip/hip_bf16.h>
#include <math.h>
#include <stdint.h>

typedef __attribute__((ext_vector_type(16))) _Float16 v16h;
typedef __attribute__((ext_vector_type(8)))  float    v8f;
typedef __attribute__((ext_vector_type(4)))  float    v4f;
typedef __attribute__((ext_vector_type(4)))  unsigned int v4u;
typedef __attribute__((ext_vector_type(8)))  int      v8i;
typedef __attribute__((ext_vector_type(4)))  int      v4i;

#define BB 1024
#define TT 512
#define KK 64
#define NEG_INF (-3.0e38f)

#if __has_builtin(__builtin_amdgcn_tensor_load_to_lds)
#define HAVE_TDM 1
#else
#define HAVE_TDM 0
#endif

// raw v_exp_f32 / v_log_f32 (skip OCML range-fixup expansions)
__device__ __forceinline__ float fast_exp(float x) {
  return __builtin_amdgcn_exp2f(x * 1.4426950408889634f);
}
__device__ __forceinline__ float fast_log(float x) {
  return __builtin_amdgcn_logf(x) * 0.6931471805599453f;
}

// ---------------------------------------------------------------------------
// prep: E = exp(trans) pre-swizzled into WMMA B-operand layout (f16).
// ---------------------------------------------------------------------------
__global__ void crf_prep_kernel(const float* __restrict__ trans,
                                _Float16* __restrict__ EB) {
  int idx = threadIdx.x + blockIdx.x * blockDim.x;  // 0..4095
  if (idx >= 8 * 32 * 16) return;
  int h    = idx & 15;
  int lane = (idx >> 4) & 31;
  int tile = idx >> 9;
  int kc   = tile >> 2;
  int jt   = tile & 3;
  int k    = kc * 32 + ((lane < 16) ? 0 : 16) + h;
  int j    = jt * 16 + (lane & 15);
  EB[idx] = (_Float16)fast_exp(trans[k * KK + j]);
}

// ---------------------------------------------------------------------------
// viterbi: one 64-thread block per sequence; thread j owns state j.
// ---------------------------------------------------------------------------
__global__ __launch_bounds__(64) void crf_viterbi_kernel(
    const float* __restrict__ logits, const int* __restrict__ nwords,
    const int* __restrict__ tags, const float* __restrict__ trans,
    unsigned char* __restrict__ bp, float* __restrict__ score,
    float* __restrict__ pred) {
  __shared__ float transS[KK * KK];
  __shared__ float alphaS[KK];
  __shared__ float redS[KK];
  __shared__ int   redI[KK];

  const int j = threadIdx.x;
  const int b = blockIdx.x;
  for (int i = j; i < KK * KK; i += KK) transS[i] = trans[i];

  const int len = nwords[b];
  const float* lg = logits + (size_t)b * TT * KK;
  unsigned char* bpb = bp + (size_t)b * (TT - 1) * KK;

  alphaS[j] = lg[j];
  __syncthreads();

  for (int t = 1; t < len; ++t) {
    __builtin_prefetch(lg + (t + 1) * KK + j, 0, 1);  // global_prefetch_b8
    float best = NEG_INF;
    int   bi   = 0;
#pragma unroll 8
    for (int i = 0; i < KK; ++i) {
      float s = alphaS[i] + transS[i * KK + j];
      if (s > best) { best = s; bi = i; }   // first-max tie-break (argmax)
    }
    float na = best + lg[t * KK + j];
    bpb[(t - 1) * KK + j] = (unsigned char)bi;
    __syncthreads();
    alphaS[j] = na;
    __syncthreads();
  }

  redS[j] = alphaS[j];
  redI[j] = j;
  __syncthreads();
#pragma unroll
  for (int off = 32; off > 0; off >>= 1) {
    if (j < off) {
      if (redS[j + off] > redS[j]) { redS[j] = redS[j + off]; redI[j] = redI[j + off]; }
    }
    __syncthreads();
  }

  if (j == 0) {
    int tag = redI[0];
    for (int t = TT - 1; t >= len - 1; --t) pred[(size_t)b * TT + t] = (float)tag;
    for (int t = len - 2; t >= 0; --t) {
      tag = bpb[t * KK + tag];
      pred[(size_t)b * TT + t] = (float)tag;
    }
  }

  float sc = 0.0f;
  for (int t = j; t < TT; t += KK) {
    int tg = tags[b * TT + t];
    if (t < len) sc += lg[t * KK + tg];
    if (t < len - 1) sc += transS[tg * KK + tags[b * TT + t + 1]];
  }
  __syncthreads();
  redS[j] = sc;
  __syncthreads();
#pragma unroll
  for (int off = 32; off > 0; off >>= 1) {
    if (j < off) redS[j] += redS[j + off];
    __syncthreads();
  }
  if (j == 0) score[b] = redS[0];
}

// ---------------------------------------------------------------------------
// lognorm: one wave32 per 16 sequences; lse-recursion as f16 WMMA GEMM.
// Double-buffered TENSOR_LOAD_TO_LDS fetches the next step's 16x64 f32
// logits tile while the current step computes.
// ---------------------------------------------------------------------------
__global__ __launch_bounds__(32) void crf_lognorm_kernel(
    const float* __restrict__ logits, const int* __restrict__ nwords,
    const _Float16* __restrict__ EB, float* __restrict__ logZ) {
  __shared__ float alphaS[16 * KK];
  __shared__ float lgS[2][16 * KK];
  __shared__ float MsS[16];

  const int lane  = threadIdx.x;
  const int lm    = lane & 15;
  const int half8 = (lane < 16) ? 0 : 8;
  const int s0    = blockIdx.x * 16;

  // Preload the 8 B-operand tiles of E = exp(trans) (64 VGPRs, resident).
  v16h Bt[2][4];
#pragma unroll
  for (int kc = 0; kc < 2; ++kc)
#pragma unroll
    for (int jt = 0; jt < 4; ++jt)
      Bt[kc][jt] = *(const v16h*)(EB + (size_t)((kc * 4 + jt) * 32 + lane) * 16);

  int len[8];
#pragma unroll
  for (int r = 0; r < 8; ++r) len[r] = nwords[s0 + r + half8];

  // wave-uniform loop bound: longest sequence in this 16-group (s_loads)
  int maxLen = 1;
#pragma unroll
  for (int i = 0; i < 16; ++i) {
    int li = nwords[s0 + i];
    maxLen = (li > maxLen) ? li : maxLen;
  }

  // alpha0 = logits[:, 0] in C-layout accumulators
  v8f acc[4];
#pragma unroll
  for (int jt = 0; jt < 4; ++jt)
#pragma unroll
    for (int r = 0; r < 8; ++r)
      acc[jt][r] = logits[(size_t)(s0 + r + half8) * TT * KK + jt * 16 + lm];

  const int mA = lm;

#if HAVE_TDM
  // Tensor DMA descriptor: 2D tensor [B rows of T*K f32], tile 64 x 16,
  // dim0 stride = T*K, data_size=4B (code 2).
  const unsigned long long gbase =
      (unsigned long long)(uintptr_t)(logits + (size_t)s0 * TT * KK);
  const unsigned int lds0 = (unsigned int)(uintptr_t)(void*)lgS[0];
  const unsigned int lds1 = (unsigned int)(uintptr_t)(void*)lgS[1];
  const v8i g1 = { 0x20000,            // data_size=2 (4B), mask=0
                   (int)0x80000000,    // tensor_dim0=32768 -> bits[63:48]
                   0x04000000,         // tensor_dim1=1024  -> bits[95:80]
                   0x00400000,         // tile_dim0=64      -> bits[127:112]
                   16,                 // tile_dim1=16
                   32768,              // tensor_dim0_stride[31:0]
                   0, 0 };
  const v4i gz4 = { 0, 0, 0, 0 };
#if defined(__clang_major__) && __clang_major__ >= 23
  const v8i gz8 = { 0, 0, 0, 0, 0, 0, 0, 0 };
#endif
  auto tdm_issue = [&](int trow, unsigned int ldsoff) {
    unsigned long long ga = gbase + (unsigned long long)trow * (KK * 4);
    v4u g0 = { 1u,                                   // count=1 (user)
               ldsoff,                               // lds_addr
               (unsigned int)ga,                     // global_addr lo
               (unsigned int)((ga >> 32) & 0x1ffffffu) | (2u << 30) };
#if defined(__clang_major__) && __clang_major__ >= 23
    __builtin_amdgcn_tensor_load_to_lds(g0, g1, gz4, gz4, gz8, 0);
#else
    __builtin_amdgcn_tensor_load_to_lds(g0, g1, gz4, gz4, 0);
#endif
  };
  if (maxLen > 1) tdm_issue(1, lds0);   // prime the pipeline
#endif

  for (int t = 1; t < maxLen; ++t) {
    const int cur = (t - 1) & 1;

    // 1) stage alpha to LDS
#pragma unroll
    for (int jt = 0; jt < 4; ++jt)
#pragma unroll
      for (int r = 0; r < 8; ++r)
        alphaS[(r + half8) * KK + jt * 16 + lm] = acc[jt][r];

    // 2) kick off DMA for the *next* step's tile (double buffer)
#if HAVE_TDM
    {
      int tp = (t + 1 < TT) ? (t + 1) : t;
      tdm_issue(tp, (t & 1) ? lds1 : lds0);
    }
#else
#pragma unroll
    for (int i = 0; i < 8; ++i) {
      int fl = lane * 4 + i * 128;          // element index in 16x64 tile
      int m  = fl >> 6, c = fl & 63;
      *(v4f*)&lgS[cur][fl] =
          *(const v4f*)&logits[(size_t)(s0 + m) * TT * KK + t * KK + c];
    }
#endif
    __syncthreads();

    // 3) per-sequence row max
    float M = NEG_INF;
#pragma unroll 8
    for (int n = 0; n < KK; ++n) M = fmaxf(M, alphaS[mA * KK + n]);
    if (lane < 16) MsS[lane] = M;
    __syncthreads();

    // 4) f16 A operands: exp(alpha - M)
    v16h A0, A1;
#pragma unroll
    for (int h = 0; h < 16; ++h) {
      int k = h + 2 * half8;
      A0[h] = (_Float16)fast_exp(alphaS[mA * KK + k] - M);
      A1[h] = (_Float16)fast_exp(alphaS[mA * KK + 32 + k] - M);
    }

    // 5) wait for *current* tile only (1 outstanding = next tile in flight)
#if HAVE_TDM
    __builtin_amdgcn_s_wait_tensorcnt(1);
#endif
    float Mm[8], lgv[4][8];
#pragma unroll
    for (int r = 0; r < 8; ++r) Mm[r] = MsS[r + half8];
#pragma unroll
    for (int jt = 0; jt < 4; ++jt)
#pragma unroll
      for (int r = 0; r < 8; ++r)
        lgv[jt][r] = lgS[cur][(r + half8) * KK + jt * 16 + lm];

    // 6) 8x v_wmma_f32_16x16x32_f16 + branchless masked update:
    //    acc += sel * (nv - acc), sel in {0,1}  (v_cmp + v_cndmask + v_fma)
    v8f vz = {};
#pragma unroll
    for (int jt = 0; jt < 4; ++jt) {
      v8f sum = __builtin_amdgcn_wmma_f32_16x16x32_f16(
          false, A0, false, Bt[0][jt], (short)0, vz, false, false);
      sum = __builtin_amdgcn_wmma_f32_16x16x32_f16(
          false, A1, false, Bt[1][jt], (short)0, sum, false, false);
#pragma unroll
      for (int r = 0; r < 8; ++r) {
        float nv  = Mm[r] + fast_log(sum[r]) + lgv[jt][r];
        float sel = (t < len[r]) ? 1.0f : 0.0f;
        acc[jt][r] = fmaf(sel, nv - acc[jt][r], acc[jt][r]);
      }
    }
    __syncthreads();
  }

  // final logZ = logsumexp over states
#pragma unroll
  for (int jt = 0; jt < 4; ++jt)
#pragma unroll
    for (int r = 0; r < 8; ++r)
      alphaS[(r + half8) * KK + jt * 16 + lm] = acc[jt][r];
  __syncthreads();
  if (lane < 16) {
    float M = NEG_INF;
    for (int n = 0; n < KK; ++n) M = fmaxf(M, alphaS[lane * KK + n]);
    float s = 0.0f;
    for (int n = 0; n < KK; ++n) s += fast_exp(alphaS[lane * KK + n] - M);
    logZ[s0 + lane] = M + fast_log(s);
  }
}

// ---------------------------------------------------------------------------
__global__ __launch_bounds__(256) void crf_loss_kernel(
    const float* __restrict__ logZ, const float* __restrict__ score,
    float* __restrict__ out) {
  __shared__ float red[256];
  int i = threadIdx.x;
  float s = 0.0f;
  for (int b = i; b < BB; b += 256) s += logZ[b] - score[b];
  red[i] = s;
  __syncthreads();
#pragma unroll
  for (int off = 128; off > 0; off >>= 1) {
    if (i < off) red[i] += red[i + off];
    __syncthreads();
  }
  if (i == 0) out[0] = red[0] / (float)BB;
}

// ---------------------------------------------------------------------------
extern "C" void kernel_launch(void* const* d_in, const int* in_sizes, int n_in,
                              void* d_out, int out_size, void* d_ws, size_t ws_size,
                              hipStream_t stream) {
  const float* logits = (const float*)d_in[0];
  const int*   nwords = (const int*)d_in[1];
  const int*   tags   = (const int*)d_in[2];
  const float* trans  = (const float*)d_in[3];
  float* out = (float*)d_out;

  char* ws = (char*)d_ws;
  _Float16*      EB    = (_Float16*)(ws);               //  8 KB
  float*         logZ  = (float*)(ws + 8192);           //  4 KB
  float*         score = (float*)(ws + 12288);          //  4 KB
  unsigned char* bp    = (unsigned char*)(ws + 16384);  // ~32 MB

  crf_prep_kernel<<<16, 256, 0, stream>>>(trans, EB);
  crf_viterbi_kernel<<<BB, 64, 0, stream>>>(logits, nwords, tags, trans, bp,
                                            score, out);
  crf_lognorm_kernel<<<BB / 16, 32, 0, stream>>>(logits, nwords, EB, logZ);
  crf_loss_kernel<<<1, 256, 0, stream>>>(logZ, score, out + (size_t)BB * TT);
}
